// SRNL_7258494730809
// MI455X (gfx1250) — matrix operand order repaired and checked by reading, not compiled, not run on hardware
//
#include <hip/hip_runtime.h>

typedef __attribute__((ext_vector_type(16))) _Float16 v16h;
typedef __attribute__((ext_vector_type(8)))  _Float16 v8h;
typedef __attribute__((ext_vector_type(4)))  _Float16 v4h;
typedef __attribute__((ext_vector_type(8)))  float    v8f;

union H16 { v16h v; v8h h[2]; };

static constexpr int N_  = 2;
static constexpr int C_  = 128;
static constexpr int HW_ = 9216;          // 96*96
static constexpr int L_  = HW_;
static constexpr int D_  = 64;            // qk dim
static constexpr int DV_ = 128;           // value dim
static constexpr int BQ_ = 128;           // query rows per block (8 waves x 16)
static constexpr int BK_ = 64;            // key tile

__device__ __forceinline__ v8f wmma16x16x32(v16h a, v16h b, v8f c) {
  // D = A(16x32 f16) * B(32x16 f16) + C(16x16 f32)
  return __builtin_amdgcn_wmma_f32_16x16x32_f16(false, a, false, b, (short)0, c,
                                                false, false);
}

// ---------------------------------------------------------------------------
// Projection kernel: y = prelu(W*x + b), emit f16 Q[L,64], K[L,64], V[L,128]
// grid = (N*HW/64, 4): y-group 0->Q(w1), 1->K(w2), 2->V[:,0:64], 3->V[:,64:128]
// ---------------------------------------------------------------------------
__global__ __launch_bounds__(256)
void proj_kernel(const float* __restrict__ x,
                 const float* __restrict__ w1, const float* __restrict__ b1, const float* __restrict__ a1,
                 const float* __restrict__ w2, const float* __restrict__ b2, const float* __restrict__ a2,
                 const float* __restrict__ wa, const float* __restrict__ ba, const float* __restrict__ aa,
                 _Float16* __restrict__ Qb, _Float16* __restrict__ Kb, _Float16* __restrict__ Vb) {
  __shared__ float sW[64 * 128];   // 32 KB
  __shared__ float sX[128 * 64];   // 32 KB, [ch][px]
  const int tid = threadIdx.x;
  const int grp = blockIdx.y;

  const float* wsrc; const float* bsrc; float slope;
  if (grp == 0)      { wsrc = w1;           bsrc = b1;      slope = a1[0]; }
  else if (grp == 1) { wsrc = w2;           bsrc = b2;      slope = a2[0]; }
  else if (grp == 2) { wsrc = wa;           bsrc = ba;      slope = aa[0]; }
  else               { wsrc = wa + 64*128;  bsrc = ba + 64; slope = aa[0]; }

  for (int i = tid; i < 64 * 128; i += 256) sW[i] = wsrc[i];

  const int g0  = blockIdx.x * 64;       // first global pixel of tile
  const int n   = g0 / HW_;              // tiles never straddle batches (64 | HW)
  const int hw0 = g0 - n * HW_;
  const float* xb = x + (size_t)n * C_ * HW_ + hw0;
  for (int i = tid; i < 128 * 64; i += 256) {
    const int ch = i >> 6, px = i & 63;
    sX[i] = xb[(size_t)ch * HW_ + px];
  }
  __syncthreads();

  const int px = tid & 63;
  const int og = tid >> 6;               // 4 groups of 16 output channels
  float acc[16];
#pragma unroll
  for (int o = 0; o < 16; ++o) acc[o] = 0.f;
  for (int ch = 0; ch < 128; ++ch) {
    const float xv = sX[ch * 64 + px];
#pragma unroll
    for (int o = 0; o < 16; ++o)
      acc[o] = fmaf(sW[(og * 16 + o) * 128 + ch], xv, acc[o]);
  }

  const int g = g0 + px;
#pragma unroll
  for (int o = 0; o < 16; ++o) {
    const int oc = og * 16 + o;
    float yv = acc[o] + bsrc[oc];
    yv = (yv >= 0.f) ? yv : slope * yv;       // PReLU (single slope)
    const _Float16 hv = (_Float16)yv;
    if (grp == 0)      Qb[(size_t)g * D_  + oc]       = hv;
    else if (grp == 1) Kb[(size_t)g * D_  + oc]       = hv;
    else if (grp == 2) Vb[(size_t)g * DV_ + oc]       = hv;
    else               Vb[(size_t)g * DV_ + 64 + oc]  = hv;
  }
}

// ---------------------------------------------------------------------------
// Flash-attention kernel (transposed tiles => lane-local softmax), double-
// buffered K/V tiles with async global->LDS copy for K (ASYNCcnt path).
// grid = (L/128, N), block = 256 (8 waves). Each wave owns 16 query columns.
//   GEMM1: S^T[key, q]  = K_tile(16k x 64d) * Q^T(64d x 16q)
//   GEMM2: O^T[dv,  q] += V^T(16dv x 64k)  * P^T(64k x 16q)
// out[n, c, q] = O^T[c, q] / l[q]
// ---------------------------------------------------------------------------
__global__ __launch_bounds__(256)
void flash_kernel(const _Float16* __restrict__ Q, const _Float16* __restrict__ K,
                  const _Float16* __restrict__ V, float* __restrict__ out) {
  __shared__ _Float16 sK[2][BK_ * D_];       // 2x 64x64 (key, dim)        16 KB
  __shared__ _Float16 sVt[2][DV_ * BK_];     // 2x 128x64 (dv, key)        32 KB
  __shared__ _Float16 sP[8 * 16 * BK_];      // per-wave P (q, key)        16 KB

  const int tid   = threadIdx.x;
  const int wave  = tid >> 5;
  const int lane  = tid & 31;
  const int lmod  = lane & 15;
  const int lhalf = lane >> 4;               // 0: lanes 0-15, 1: lanes 16-31
  const int n     = blockIdx.y;
  const int qw    = blockIdx.x * BQ_ + wave * 16;   // wave's query base

  const _Float16* Qg = Q + (size_t)n * L_ * D_;
  const _Float16* Kg = K + (size_t)n * L_ * D_;
  const _Float16* Vg = V + (size_t)n * L_ * DV_;

  const int abase = lhalf ? 16 : 0;          // A-fragment dim/key base (contiguous layout)
  const int kb16  = lhalf ? 16 : 0;          // B-fragment contraction base

  // Q^T B-fragments (persistent): b[e] = Q[(qw+lmod)*64 + c*32 + kb16 + e]
  H16 qb[2];
#pragma unroll
  for (int c = 0; c < 2; ++c) {
    const _Float16* src = Qg + (size_t)(qw + lmod) * D_ + c * 32 + kb16;
    qb[c].h[0] = *(const v8h*)(src);
    qb[c].h[1] = *(const v8h*)(src + 8);
  }

  v8f oacc[8];
#pragma unroll
  for (int d = 0; d < 8; ++d) oacc[d] = (v8f){0.f,0.f,0.f,0.f,0.f,0.f,0.f,0.f};
  float mrun = -3.0e38f, lrun = 0.f;

  _Float16* myP = sP + wave * (16 * BK_);

  // cooperative-load index precompute
  const int iK  = tid * 16;                  // K tile: 4096 halves, 32B/thread
  const int dv0 = (tid & 15) * 8;            // V tile: 8 dv x 4 keys per thread
  const int k0  = (tid >> 4) * 4;

  // ---- prologue: fill buffer 0 ----
  {
    const unsigned ldsK = (unsigned)(uintptr_t)&sK[0][iK];   // low 32b = LDS offset
    const _Float16* gK  = &Kg[iK];
    asm volatile("global_load_async_to_lds_b128 %0, %1, off"
                 :: "v"(ldsK), "v"(gK) : "memory");
    asm volatile("global_load_async_to_lds_b128 %0, %1, off offset:16"
                 :: "v"(ldsK), "v"(gK) : "memory");
    v8h vr[4];
#pragma unroll
    for (int kk = 0; kk < 4; ++kk)
      vr[kk] = *(const v8h*)&Vg[(size_t)(k0 + kk) * DV_ + dv0];
#pragma unroll
    for (int j = 0; j < 8; ++j) {
      v4h pk = { vr[0][j], vr[1][j], vr[2][j], vr[3][j] };
      *(v4h*)&sVt[0][(dv0 + j) * BK_ + k0] = pk;
    }
    asm volatile("s_wait_asynccnt 0x0" ::: "memory");
  }
  __syncthreads();

  int ibuf = 0;
  for (int kb = 0; kb < L_; kb += BK_, ibuf ^= 1) {
    const int nbuf = ibuf ^ 1;
    const bool hasNext = (kb + BK_) < L_;

    // ---- start next tile: async K copy + V global loads (overlap compute) ----
    v8h vr[4];
    if (hasNext) {
      const unsigned ldsK = (unsigned)(uintptr_t)&sK[nbuf][iK];
      const _Float16* gK  = &Kg[(size_t)(kb + BK_) * D_ + iK];
      asm volatile("global_load_async_to_lds_b128 %0, %1, off"
                   :: "v"(ldsK), "v"(gK) : "memory");
      asm volatile("global_load_async_to_lds_b128 %0, %1, off offset:16"
                   :: "v"(ldsK), "v"(gK) : "memory");
#pragma unroll
      for (int kk = 0; kk < 4; ++kk)
        vr[kk] = *(const v8h*)&Vg[(size_t)(kb + BK_ + k0 + kk) * DV_ + dv0];
      __builtin_prefetch(&Kg[(size_t)(kb + 2 * BK_) * D_ + iK], 0, 1);
      __builtin_prefetch(&Vg[(size_t)(kb + 2 * BK_ + k0) * DV_ + dv0], 0, 1);
    }

    // ---- S^T = K * Q^T : four 16x16 tiles (keys as rows) ----
    const _Float16* sKb = sK[ibuf];
    v8f st[4];
#pragma unroll
    for (int kt = 0; kt < 4; ++kt) {
      v8f s = (v8f){0.f,0.f,0.f,0.f,0.f,0.f,0.f,0.f};
#pragma unroll
      for (int c = 0; c < 2; ++c) {
        H16 af;  // A frag: key row (kt*16 + lmod), dims with contiguous layout
        const _Float16* as = &sKb[(size_t)(kt * 16 + lmod) * D_ + c * 32 + abase];
        af.h[0] = *(const v8h*)(as);
        af.h[1] = *(const v8h*)(as + 8);
        s = wmma16x16x32(af.v, qb[c].v, s);
      }
      st[kt] = s;
    }

    // ---- lane-local online softmax over keys ----
    float mx = st[0][0];
#pragma unroll
    for (int kt = 0; kt < 4; ++kt)
#pragma unroll
      for (int r = 0; r < 8; ++r) mx = fmaxf(mx, st[kt][r]);
    mx = fmaxf(mx, __shfl_xor(mx, 16, 32));          // partner half's 32 keys
    const float mnew = fmaxf(mrun, mx);
    const float alpha = __expf(mrun - mnew);
    float ps = 0.f;
#pragma unroll
    for (int kt = 0; kt < 4; ++kt) {
      v8h pk;
#pragma unroll
      for (int r = 0; r < 8; ++r) {
        const float p = __expf(st[kt][r] - mnew);
        ps += p;
        pk[r] = (_Float16)p;
      }
      // sP[q][key]: keys kt*16 + lhalf*8 + (0..7), contiguous 16B store
      *(v8h*)&myP[lmod * BK_ + kt * 16 + lhalf * 8] = pk;
    }
    ps += __shfl_xor(ps, 16, 32);
    lrun = alpha * lrun + ps;
    mrun = mnew;
#pragma unroll
    for (int d = 0; d < 8; ++d)
#pragma unroll
      for (int r = 0; r < 8; ++r) oacc[d][r] *= alpha;

    // wave-local LDS write->read of P (in-order DS pipe; drain DScnt)
    asm volatile("s_wait_dscnt 0x0" ::: "memory");

    // P^T B-fragments (shared by all 8 dv tiles): b[e] = P[q=lmod][c*32+kb16+e]
    H16 pb[2];
#pragma unroll
    for (int c = 0; c < 2; ++c) {
      const _Float16* psrc = &myP[lmod * BK_ + c * 32 + kb16];
      pb[c].h[0] = *(const v8h*)(psrc);
      pb[c].h[1] = *(const v8h*)(psrc + 8);
    }

    // ---- O^T += V^T * P^T : 8 dv tiles x 2 contraction chunks ----
    const _Float16* sVb = sVt[ibuf];
#pragma unroll
    for (int d = 0; d < 8; ++d) {
#pragma unroll
      for (int c = 0; c < 2; ++c) {
        H16 af;  // A frag: dv row (d*16 + lmod), keys with contiguous layout
        const _Float16* as = &sVb[(size_t)(d * 16 + lmod) * BK_ + c * 32 + abase];
        af.h[0] = *(const v8h*)(as);
        af.h[1] = *(const v8h*)(as + 8);
        oacc[d] = wmma16x16x32(af.v, pb[c].v, oacc[d]);
      }
    }

    // ---- finish next tile: transpose-store V, drain async K, publish ----
    if (hasNext) {
#pragma unroll
      for (int j = 0; j < 8; ++j) {
        v4h pk = { vr[0][j], vr[1][j], vr[2][j], vr[3][j] };
        *(v4h*)&sVt[nbuf][(dv0 + j) * BK_ + k0] = pk;
      }
      asm volatile("s_wait_asynccnt 0x0" ::: "memory");
    }
    __syncthreads();   // single barrier per tile (double-buffered)
  }

  // ---- epilogue: out[n, c, q] = O^T[c, q] / l[q] (l is lane-local) ----
  float* ob = out + (size_t)n * DV_ * L_;
  const float inv_l = 1.f / lrun;
  const int q = qw + lmod;
#pragma unroll
  for (int d = 0; d < 8; ++d) {
#pragma unroll
    for (int r = 0; r < 8; ++r) {
      const int cch = d * 16 + r + 8 * lhalf;
      ob[(size_t)cch * L_ + q] = oacc[d][r] * inv_l;
    }
  }
}

// ---------------------------------------------------------------------------
extern "C" void kernel_launch(void* const* d_in, const int* in_sizes, int n_in,
                              void* d_out, int out_size, void* d_ws, size_t ws_size,
                              hipStream_t stream) {
  const float* x  = (const float*)d_in[0];
  const float* w1 = (const float*)d_in[1];
  const float* b1 = (const float*)d_in[2];
  const float* a1 = (const float*)d_in[3];
  const float* w2 = (const float*)d_in[4];
  const float* b2 = (const float*)d_in[5];
  const float* a2 = (const float*)d_in[6];
  const float* wa = (const float*)d_in[7];
  const float* ba = (const float*)d_in[8];
  const float* aa = (const float*)d_in[9];

  _Float16* Qb = (_Float16*)d_ws;                       // [N*L, 64]
  _Float16* Kb = Qb + (size_t)N_ * L_ * D_;             // [N*L, 64]
  _Float16* Vb = Kb + (size_t)N_ * L_ * D_;             // [N*L, 128]

  proj_kernel<<<dim3((N_ * HW_) / 64, 4), 256, 0, stream>>>(
      x, w1, b1, a1, w2, b2, a2, wa, ba, aa, Qb, Kb, Vb);

  flash_kernel<<<dim3(L_ / BQ_, N_), 256, 0, stream>>>(
      Qb, Kb, Vb, (float*)d_out);
}
